// SMILESDecoder_53858889892506
// MI455X (gfx1250) — compile-verified, matrix operand order checked by reading, not run on hardware
//
#include <hip/hip_runtime.h>
#include <math.h>

#define B_  16
#define S_  512
#define V_  512
#define H_  1024
#define NL_ 8
#define NH_ 16
#define DH_ 64
#define F_  4096

typedef __attribute__((ext_vector_type(16))) __bf16         v16bf;
typedef __attribute__((ext_vector_type(8)))  unsigned int   v8u;
typedef __attribute__((ext_vector_type(8)))  float          v8f;
typedef __attribute__((ext_vector_type(4)))  int            i4;

#if __has_builtin(__builtin_amdgcn_global_load_async_to_lds_b128)
#define HAS_ASYNC 1
#endif

static __device__ __forceinline__ unsigned short f2bf(float f) {
  unsigned int u = __builtin_bit_cast(unsigned int, f);
  unsigned int r = u + 0x7FFFu + ((u >> 16) & 1u);  // round-to-nearest-even
  return (unsigned short)(r >> 16);
}

// K index of pair p (0..7) for a 16-bit 16x32 WMMA A/B fragment, lane-half g.
// ISA: VGPR0..3 -> K={0,2,4,6}+8g (+1), VGPR4..7 -> K={16,18,20,22}+8g (+1).
static __device__ __forceinline__ int kmap(int p, int g) {
  return ((p < 4) ? (2 * p) : (16 + 2 * (p - 4))) + 8 * g;
}

// 16-byte global->LDS copy: async (ASYNCcnt-tracked) on gfx1250.
static __device__ __forceinline__ void cp16(const unsigned short* g,
                                            unsigned short* l) {
#ifdef HAS_ASYNC
  __builtin_amdgcn_global_load_async_to_lds_b128(
      (__attribute__((address_space(1))) i4*)(unsigned long long)(const void*)g,
      (__attribute__((address_space(3))) i4*)(unsigned int)(unsigned long long)(void*)l,
      0, 0);
#else
  *(uint4*)l = *(const uint4*)g;
#endif
}

static __device__ __forceinline__ void wait_cp() {
#ifdef HAS_ASYNC
#if __has_builtin(__builtin_amdgcn_s_wait_asynccnt)
  __builtin_amdgcn_s_wait_asynccnt(0);
#else
  asm volatile("s_wait_asynccnt 0x0" ::: "memory");
#endif
#endif
}

// ---------------------------------------------------------------------------
// bf16 WMMA GEMM: C = epi(A[MxK]_bf16 * Wt[NxK]_bf16 + bias [, res])
// EPI: 0 = +bias -> bf16, 1 = +bias+residual -> f32, 2 = +bias+gelu -> bf16,
//      3 = plain -> f32
// ---------------------------------------------------------------------------
template <int EPI>
__global__ __launch_bounds__(256) void gemm_wmma(
    const unsigned short* __restrict__ A, const unsigned short* __restrict__ Wt,
    const float* __restrict__ bias, const float* __restrict__ res,
    void* __restrict__ Cv, int M, int N, int K) {
  __shared__ unsigned short As[128 * 40];  // [row][k] stride 80B (16B-aligned chunks)
  __shared__ unsigned short Bs[128 * 40];  // [n][k]
  const int tid = threadIdx.x;
  const int wave = tid >> 5, lane = tid & 31;
  const int g = lane >> 4, lc = lane & 15;
  const int wm = wave >> 2, wn = wave & 3;  // 2x4 wave grid -> 64x32 per wave
  const int m0 = blockIdx.y * 128, n0 = blockIdx.x * 128;

  v8f acc[4][2];
#pragma unroll
  for (int i = 0; i < 4; i++)
#pragma unroll
    for (int j = 0; j < 2; j++) acc[i][j] = (v8f)0.f;

  for (int k0 = 0; k0 < K; k0 += 32) {
    __syncthreads();
#pragma unroll
    for (int i = 0; i < 2; i++) {  // A tile: 128x32 bf16 = 512 x 16B chunks
      int idx = i * 256 + tid;
      int row = idx >> 2, c = (idx & 3) << 3;
      cp16(A + (size_t)(m0 + row) * K + k0 + c, &As[row * 40 + c]);
    }
#pragma unroll
    for (int i = 0; i < 2; i++) {  // B tile: 128n x 32k
      int idx = i * 256 + tid;
      int n = idx >> 2, c = (idx & 3) << 3;
      cp16(Wt + (size_t)(n0 + n) * K + k0 + c, &Bs[n * 40 + c]);
    }
    wait_cp();
    __syncthreads();

    v8u bf[2];
#pragma unroll
    for (int nt = 0; nt < 2; nt++) {
      const int nrow = wn * 32 + nt * 16 + lc;
#pragma unroll
      for (int p = 0; p < 8; p++)
        bf[nt][p] = *(const unsigned int*)&Bs[nrow * 40 + kmap(p, g)];
    }
#pragma unroll
    for (int mt = 0; mt < 4; mt++) {
      const int arow = wm * 64 + mt * 16 + lc;
      v8u af;
#pragma unroll
      for (int p = 0; p < 8; p++)
        af[p] = *(const unsigned int*)&As[arow * 40 + kmap(p, g)];
      v16bf a = __builtin_bit_cast(v16bf, af);
#pragma unroll
      for (int nt = 0; nt < 2; nt++) {
        acc[mt][nt] = __builtin_amdgcn_wmma_f32_16x16x32_bf16(
            false, a, false, __builtin_bit_cast(v16bf, bf[nt]), (short)0,
            acc[mt][nt], false, false);
      }
    }
  }

#pragma unroll
  for (int mt = 0; mt < 4; mt++) {
#pragma unroll
    for (int nt = 0; nt < 2; nt++) {
      int n = n0 + wn * 32 + nt * 16 + lc;
      float bv = 0.f;
      if (EPI != 3) bv = bias[n];
#pragma unroll
      for (int r = 0; r < 8; r++) {
        int m = m0 + wm * 64 + mt * 16 + r + 8 * g;
        float v = acc[mt][nt][r] + bv;
        if (EPI == 1) v += res[(size_t)m * N + n];
        if (EPI == 2) {
          float t = v;
          v = 0.5f * t * (1.f + tanhf(0.7978845608f * (t + 0.044715f * t * t * t)));
        }
        if (EPI == 0 || EPI == 2)
          ((unsigned short*)Cv)[(size_t)m * N + n] = f2bf(v);
        else
          ((float*)Cv)[(size_t)m * N + n] = v;
      }
    }
  }
}

// ---------------------------------------------------------------------------
// Flash attention (bf16 qkv) with ALiBi + causal + split-block mask.
// grid: B*NH*(S/128); block: 256 (8 waves, each owns a 16-row q tile)
// ---------------------------------------------------------------------------
__global__ __launch_bounds__(256) void attn_wmma(
    const unsigned short* __restrict__ qkv, const int* __restrict__ splits,
    unsigned short* __restrict__ out) {
  __shared__ unsigned short Ks[32 * 72];     // [key][dh] stride 144B
  __shared__ unsigned short Vt[64 * 40];     // [dh][key] stride 80B
  __shared__ unsigned short Ps[8][16 * 34];  // per-wave P scratch [q][key]
  const int tid = threadIdx.x;
  const int wave = tid >> 5, lane = tid & 31;
  const int g = lane >> 4, lc = lane & 15;
  const int bid = blockIdx.x;
  const int qb = bid & 3;         // S/128 = 4
  const int h = (bid >> 2) & 15;  // NH
  const int b = bid >> 6;
  const int q0 = qb * 128 + wave * 16;
  const int sp = splits[b];
  const float slope = exp2f(-0.5f * (float)(h + 1));

  // Preload Q fragments directly from bf16 global (pairs are contiguous)
  v8u qf[2];
  const size_t qbase = (size_t)(b * S_ + q0 + lc) * (3 * H_) + h * DH_;
#pragma unroll
  for (int f = 0; f < 2; f++)
#pragma unroll
    for (int p = 0; p < 8; p++)
      qf[f][p] = *(const unsigned int*)&qkv[qbase + 32 * f + kmap(p, g)];

  float mr[8], lr[8];
#pragma unroll
  for (int r = 0; r < 8; r++) { mr[r] = -3e38f; lr[r] = 0.f; }
  v8f o[4];
#pragma unroll
  for (int t = 0; t < 4; t++) o[t] = (v8f)0.f;

  for (int kc = 0; kc < S_ / 32; kc++) {
    const int kk0 = kc * 32;
    __syncthreads();
    {  // K chunk: 32 keys x 64 dh bf16 = 256 x 16B chunks (async)
      int key = tid >> 3, c = (tid & 7) << 3;
      size_t base = (size_t)(b * S_ + kk0 + key) * (3 * H_) + H_ + h * DH_;
      cp16(qkv + base + c, &Ks[key * 72 + c]);
    }
#pragma unroll
    for (int i = 0; i < 4; i++) {  // V chunk transposed into Vt[dh][key]
      int idx = i * 256 + tid;
      int key = idx >> 5, dh = (idx & 31) << 1;
      size_t base = (size_t)(b * S_ + kk0 + key) * (3 * H_) + 2 * H_ + h * DH_;
      unsigned int u = *(const unsigned int*)&qkv[base + dh];
      Vt[dh * 40 + key] = (unsigned short)u;
      Vt[(dh + 1) * 40 + key] = (unsigned short)(u >> 16);
    }
    wait_cp();
    __syncthreads();

    float sc[2][8];
#pragma unroll
    for (int s = 0; s < 2; s++) {  // two 16-key subtiles
      v8f a = (v8f)0.f;
#pragma unroll
      for (int f = 0; f < 2; f++) {
        v8u bfr;
#pragma unroll
        for (int p = 0; p < 8; p++)
          bfr[p] = *(const unsigned int*)&Ks[(s * 16 + lc) * 72 + 32 * f + kmap(p, g)];
        a = __builtin_amdgcn_wmma_f32_16x16x32_bf16(
            false, __builtin_bit_cast(v16bf, qf[f]), false,
            __builtin_bit_cast(v16bf, bfr), (short)0, a, false, false);
      }
      const int j = kk0 + s * 16 + lc;
#pragma unroll
      for (int r = 0; r < 8; r++) {
        int qi = q0 + r + 8 * g;
        bool allowed = (j <= qi) && !((qi >= sp) && (j < sp));
        sc[s][r] = allowed ? (0.125f * a[r] - slope * (float)(qi - j)) : -1e9f;
      }
    }

    // online softmax (row reductions across the 16 col-lanes of each half)
#pragma unroll
    for (int r = 0; r < 8; r++) {
      float rm = fmaxf(sc[0][r], sc[1][r]);
      for (int msk = 1; msk < 16; msk <<= 1) rm = fmaxf(rm, __shfl_xor(rm, msk, 32));
      float mnew = fmaxf(mr[r], rm);
      float fr = __expf(mr[r] - mnew);
      mr[r] = mnew;
      sc[0][r] = __expf(sc[0][r] - mnew);
      sc[1][r] = __expf(sc[1][r] - mnew);
      float rs = sc[0][r] + sc[1][r];
      for (int msk = 1; msk < 16; msk <<= 1) rs += __shfl_xor(rs, msk, 32);
      lr[r] = lr[r] * fr + rs;
#pragma unroll
      for (int t = 0; t < 4; t++) o[t][r] *= fr;
    }

    // C-layout -> A-layout relayout of P through per-wave LDS scratch
#pragma unroll
    for (int s = 0; s < 2; s++)
#pragma unroll
      for (int r = 0; r < 8; r++)
        Ps[wave][(r + 8 * g) * 34 + s * 16 + lc] = f2bf(sc[s][r]);
    __builtin_amdgcn_wave_barrier();

    v8u pf;
#pragma unroll
    for (int p = 0; p < 8; p++)
      pf[p] = *(const unsigned int*)&Ps[wave][lc * 34 + kmap(p, g)];
#pragma unroll
    for (int t = 0; t < 4; t++) {  // O += P @ V
      v8u bfr;
#pragma unroll
      for (int p = 0; p < 8; p++)
        bfr[p] = *(const unsigned int*)&Vt[(t * 16 + lc) * 40 + kmap(p, g)];
      o[t] = __builtin_amdgcn_wmma_f32_16x16x32_bf16(
          false, __builtin_bit_cast(v16bf, pf), false,
          __builtin_bit_cast(v16bf, bfr), (short)0, o[t], false, false);
    }
  }

#pragma unroll
  for (int r = 0; r < 8; r++) {
    float inv = 1.f / lr[r];
    int m = b * S_ + q0 + r + 8 * g;
#pragma unroll
    for (int t = 0; t < 4; t++)
      out[(size_t)m * H_ + h * DH_ + t * 16 + lc] = f2bf(o[t][r] * inv);
  }
}

// ---------------------------------------------------------------------------
__global__ __launch_bounds__(256) void layernorm_k(
    const float* __restrict__ x, const float* __restrict__ gw,
    const float* __restrict__ bw, unsigned short* __restrict__ y) {
  __shared__ float red[256];
  const int row = blockIdx.x;
  const float* xr = x + (size_t)row * H_;
  float vals[4];
  float lsum = 0.f;
#pragma unroll
  for (int i = 0; i < 4; i++) { vals[i] = xr[threadIdx.x + i * 256]; lsum += vals[i]; }
  red[threadIdx.x] = lsum;
  __syncthreads();
  for (int s = 128; s > 0; s >>= 1) {
    if (threadIdx.x < s) red[threadIdx.x] += red[threadIdx.x + s];
    __syncthreads();
  }
  float mu = red[0] * (1.f / H_);
  __syncthreads();
  float lvar = 0.f;
#pragma unroll
  for (int i = 0; i < 4; i++) { float d = vals[i] - mu; lvar += d * d; }
  red[threadIdx.x] = lvar;
  __syncthreads();
  for (int s = 128; s > 0; s >>= 1) {
    if (threadIdx.x < s) red[threadIdx.x] += red[threadIdx.x + s];
    __syncthreads();
  }
  float rs = rsqrtf(red[0] * (1.f / H_) + 1e-5f);
#pragma unroll
  for (int i = 0; i < 4; i++) {
    int c = threadIdx.x + i * 256;
    y[(size_t)row * H_ + c] = f2bf((vals[i] - mu) * rs * gw[c] + bw[c]);
  }
}

__global__ __launch_bounds__(256) void embed_k(const int* __restrict__ ids,
                                               const float* __restrict__ emb,
                                               float* __restrict__ x) {
  int row = blockIdx.x;
  int id = ids[row];
  const float4* src = (const float4*)(emb + (size_t)id * H_);
  float4* dst = (float4*)(x + (size_t)row * H_);
  dst[threadIdx.x] = src[threadIdx.x];
}

// Convert + transpose fp32 W[K,N] -> bf16 Wt[N,K]. K = 1<<kshift.
__global__ __launch_bounds__(256) void convt_k(const float* __restrict__ W,
                                               unsigned short* __restrict__ Wt,
                                               int N, int kshift) {
  unsigned int i = blockIdx.x * 256 + threadIdx.x;
  int k = i & ((1u << kshift) - 1u);
  int n = i >> kshift;
  Wt[i] = f2bf(W[(size_t)k * N + n]);
}

// Contiguous fp32 -> bf16 convert (for emb, already [N,K]).
__global__ __launch_bounds__(256) void conv_k(const float* __restrict__ W,
                                              unsigned short* __restrict__ Wt) {
  unsigned int i = blockIdx.x * 256 + threadIdx.x;
  Wt[i] = f2bf(W[i]);
}

// ---------------------------------------------------------------------------
extern "C" void kernel_launch(void* const* d_in, const int* in_sizes, int n_in,
                              void* d_out, int out_size, void* d_ws,
                              size_t ws_size, hipStream_t stream) {
  const int*   ids    = (const int*)d_in[0];
  const int*   splits = (const int*)d_in[1];
  const float* emb    = (const float*)d_in[2];
  const float* ln1_g  = (const float*)d_in[3];
  const float* ln1_b  = (const float*)d_in[4];
  const float* Wqkv   = (const float*)d_in[5];
  const float* bqkv   = (const float*)d_in[6];
  const float* Wo     = (const float*)d_in[7];
  const float* bo     = (const float*)d_in[8];
  const float* ln2_g  = (const float*)d_in[9];
  const float* ln2_b  = (const float*)d_in[10];
  const float* Wfc    = (const float*)d_in[11];
  const float* bfc    = (const float*)d_in[12];
  const float* Wproj  = (const float*)d_in[13];
  const float* bproj  = (const float*)d_in[14];
  const float* lnf_g  = (const float*)d_in[15];
  const float* lnf_b  = (const float*)d_in[16];

  const int M = B_ * S_;
  typedef unsigned short u16;
  char* ws = (char*)d_ws;
  size_t off = 0;
  float* x    = (float*)(ws + off); off += (size_t)M * H_ * 4;
  u16* hbuf   = (u16*)(ws + off);   off += (size_t)M * H_ * 2;
  u16* qkvb   = (u16*)(ws + off);   off += (size_t)M * 3 * H_ * 2;
  u16* attnb  = (u16*)(ws + off);   off += (size_t)M * H_ * 2;
  u16* ffnb   = (u16*)(ws + off);   off += (size_t)M * F_ * 2;
  u16* wq_t   = (u16*)(ws + off);   off += (size_t)NL_ * 3 * H_ * H_ * 2;
  u16* wo_t   = (u16*)(ws + off);   off += (size_t)NL_ * H_ * H_ * 2;
  u16* wfc_t  = (u16*)(ws + off);   off += (size_t)NL_ * F_ * H_ * 2;
  u16* wp_t   = (u16*)(ws + off);   off += (size_t)NL_ * H_ * F_ * 2;
  u16* embb   = (u16*)(ws + off);   off += (size_t)V_ * H_ * 2;

  // One-time (per launch) weight convert + transpose to bf16 [N,K]
  for (int l = 0; l < NL_; l++) {
    convt_k<<<(3 * H_ * H_) / 256, 256, 0, stream>>>(
        Wqkv + (size_t)l * H_ * 3 * H_, wq_t + (size_t)l * 3 * H_ * H_, 3 * H_, 10);
    convt_k<<<(H_ * H_) / 256, 256, 0, stream>>>(
        Wo + (size_t)l * H_ * H_, wo_t + (size_t)l * H_ * H_, H_, 10);
    convt_k<<<(H_ * F_) / 256, 256, 0, stream>>>(
        Wfc + (size_t)l * H_ * F_, wfc_t + (size_t)l * F_ * H_, F_, 10);
    convt_k<<<(F_ * H_) / 256, 256, 0, stream>>>(
        Wproj + (size_t)l * F_ * H_, wp_t + (size_t)l * H_ * F_, H_, 12);
  }
  conv_k<<<(V_ * H_) / 256, 256, 0, stream>>>(emb, embb);

  embed_k<<<M, 256, 0, stream>>>(ids, emb, x);

  for (int l = 0; l < NL_; l++) {
    layernorm_k<<<M, 256, 0, stream>>>(x, ln1_g + (size_t)l * H_,
                                       ln1_b + (size_t)l * H_, hbuf);
    gemm_wmma<0><<<dim3(3 * H_ / 128, M / 128), 256, 0, stream>>>(
        hbuf, wq_t + (size_t)l * 3 * H_ * H_, bqkv + (size_t)l * 3 * H_,
        nullptr, qkvb, M, 3 * H_, H_);
    attn_wmma<<<B_ * NH_ * (S_ / 128), 256, 0, stream>>>(qkvb, splits, attnb);
    gemm_wmma<1><<<dim3(H_ / 128, M / 128), 256, 0, stream>>>(
        attnb, wo_t + (size_t)l * H_ * H_, bo + (size_t)l * H_, x, x, M, H_, H_);
    layernorm_k<<<M, 256, 0, stream>>>(x, ln2_g + (size_t)l * H_,
                                       ln2_b + (size_t)l * H_, hbuf);
    gemm_wmma<2><<<dim3(F_ / 128, M / 128), 256, 0, stream>>>(
        hbuf, wfc_t + (size_t)l * F_ * H_, bfc + (size_t)l * F_, nullptr, ffnb,
        M, F_, H_);
    gemm_wmma<1><<<dim3(H_ / 128, M / 128), 256, 0, stream>>>(
        ffnb, wp_t + (size_t)l * H_ * F_, bproj + (size_t)l * H_, x, x, M, H_,
        F_);
  }

  layernorm_k<<<M, 256, 0, stream>>>(x, lnf_g, lnf_b, hbuf);
  gemm_wmma<3><<<dim3(V_ / 128, M / 128), 256, 0, stream>>>(
      hbuf, embb, nullptr, nullptr, d_out, M, V_, H_);
}
